// iTPN_CLIP_61005715472427
// MI455X (gfx1250) — matrix operand order repaired and checked by reading, not compiled, not run on hardware
//
#include <hip/hip_runtime.h>
#include <hip/hip_bf16.h>

// B=2 H=48 W=48 DIM=768 NH=12 HD=64 -> SEQ=2304, 24 global heads, 4608 rows.
#define NHEADS 24
#define SEQ    2304
#define HDV    64
#define DIMV   768
#define MROWS  4608

typedef __bf16 v16bf __attribute__((ext_vector_type(16)));
typedef float  v8f   __attribute__((ext_vector_type(8)));
typedef unsigned int u32x4 __attribute__((ext_vector_type(4)));
typedef int          i32x8 __attribute__((ext_vector_type(8)));
typedef int          i32x4 __attribute__((ext_vector_type(4)));

#if defined(__has_builtin)
#  if __has_builtin(__builtin_amdgcn_tensor_load_to_lds) && \
      __has_builtin(__builtin_amdgcn_s_wait_tensorcnt)
#    define HAVE_TDM 1
#  else
#    define HAVE_TDM 0
#  endif
#else
#  define HAVE_TDM 0
#endif

// A-fragment (16x32 bf16) K index for element e in lane-half hl (lane>>4):
// one contiguous 16-element run at offset 16*hl.
__device__ __forceinline__ int a_k_idx(int e, int hl) {
    return e + (hl << 4);
}

__device__ __forceinline__ v8f wmma_bf16(v16bf a, v16bf b, v8f c) {
    return __builtin_amdgcn_wmma_f32_16x16x32_bf16(false, a, false, b,
                                                   (short)0, c, false, false);
}

__device__ __forceinline__ unsigned lds_addr_of(const void* p) {
    // generic LDS pointer: addr[31:0] is the LDS byte offset (CDNA5 aperture)
    return (unsigned)(unsigned long long)p;
}

#if HAVE_TDM
// 2D bf16 tile DMA: width elems contiguous per row, `rows` rows, row stride
// `stride_elems` (all in 2-byte units). D# groups per CDNA5 ISA ch.8.
// This toolchain exposes the 6-arg builtin (extra i32x8 group, zero-filled).
__device__ __forceinline__ void tdm_load_2d_bf16(
    unsigned lds_addr, const void* gptr, unsigned width, unsigned rows,
    unsigned long long stride_elems)
{
    unsigned long long ga = (unsigned long long)gptr;
    u32x4 g0;
    g0[0] = 1u;                                                 // count=1, user
    g0[1] = lds_addr;                                           // lds_addr
    g0[2] = (unsigned)(ga & 0xffffffffull);                     // global[31:0]
    g0[3] = (unsigned)((ga >> 32) & 0x01ffffffull) | (2u << 30);// global[56:32], type=2
    i32x8 g1;
    g1[0] = 1 << 16;                                            // data_size=1 (2B)
    g1[1] = (int)((width & 0xffffu) << 16);                     // tensor_dim0[15:0]
    g1[2] = (int)((width >> 16) & 0xffffu) | (int)((rows & 0xffffu) << 16);
    g1[3] = (int)((rows >> 16) & 0xffffu) | (int)((width & 0xffffu) << 16); // tile_dim0
    g1[4] = (int)(rows & 0xffffu);                              // tile_dim1, tile_dim2=0
    g1[5] = (int)(stride_elems & 0xffffffffull);                // dim0_stride[31:0]
    g1[6] = (int)((stride_elems >> 32) & 0xffffull);            // dim0_stride[47:32]
    g1[7] = 0;
    i32x4 gz4 = {};
    i32x8 gz8 = {};
    __builtin_amdgcn_tensor_load_to_lds(g0, g1, gz4, gz4, gz8, 0);
}
#endif

// ---------------------------------------------------------------------------
// Kernel 0: f32 -> bf16 pack (n must be a multiple of 8)
// ---------------------------------------------------------------------------
__global__ void __launch_bounds__(256) pack_bf16_kernel(
    const float* __restrict__ src, __bf16* __restrict__ dst, int n)
{
    int i = (blockIdx.x * 256 + threadIdx.x) * 8;
    if (i + 8 <= n) {
#pragma unroll
        for (int j = 0; j < 8; ++j) dst[i + j] = (__bf16)src[i + j];
    }
}

// ---------------------------------------------------------------------------
// Kernel 1: QKV projection GEMM (4608x768)@(768x2304)+bias, wave tile 32x64.
// Shared B tile (block-uniform 64 cols) staged via TDM into double-buffered
// LDS; A fragments direct from global. One head / one of q,k,v per wave tile.
// ---------------------------------------------------------------------------
__global__ void __launch_bounds__(256) qkv_gemm_kernel(
    const __bf16* __restrict__ xb, const __bf16* __restrict__ wb,
    const float* __restrict__ qkv_b, __bf16* __restrict__ qbuf,
    __bf16* __restrict__ kbuf, __bf16* __restrict__ vT)
{
    __shared__ __bf16 ldsB[2][64][32];
    const int lane = threadIdx.x & 31, wave = threadIdx.x >> 5;
    const int hl = lane >> 4, n16 = lane & 15;
    const int m0 = blockIdx.x * 256 + wave * 32;
    const int nb = blockIdx.y * 64;
    v8f acc[2][4] = {};

    auto stageB = [&](int bb, int kc) {
#if HAVE_TDM
        if (wave == 0)
            tdm_load_2d_bf16(lds_addr_of(&ldsB[bb][0][0]),
                             &wb[(size_t)nb * DIMV + kc], 32, 64, DIMV);
#else
        int brow = threadIdx.x >> 2, bcol = (threadIdx.x & 3) * 8;
        const __bf16* bs = &wb[(size_t)(nb + brow) * DIMV + kc + bcol];
#pragma unroll
        for (int j = 0; j < 8; ++j) ldsB[bb][brow][bcol + j] = bs[j];
#endif
    };
    auto commit = [&]() {
#if HAVE_TDM
        if (wave == 0) __builtin_amdgcn_s_wait_tensorcnt((short)0);
#endif
        __syncthreads();
    };

    stageB(0, 0);
    commit();
    for (int c = 0; c < DIMV / 32; ++c) {
        const int kc = c * 32, bb = c & 1;
        if (kc + 32 < DIMV) stageB(bb ^ 1, kc + 32);
        v16bf af[2], bfr[4];
#pragma unroll
        for (int mi = 0; mi < 2; ++mi) {
            const __bf16* ap = &xb[(size_t)(m0 + mi * 16 + n16) * DIMV + kc];
#pragma unroll
            for (int e = 0; e < 16; ++e) af[mi][e] = ap[a_k_idx(e, hl)];
        }
#pragma unroll
        for (int ni = 0; ni < 4; ++ni) {
            const __bf16* bp = &ldsB[bb][ni * 16 + n16][16 * hl];
#pragma unroll
            for (int e = 0; e < 16; ++e) bfr[ni][e] = bp[e];
        }
#pragma unroll
        for (int mi = 0; mi < 2; ++mi)
#pragma unroll
            for (int ni = 0; ni < 4; ++ni)
                acc[mi][ni] = wmma_bf16(af[mi], bfr[ni], acc[mi][ni]);
        commit();
    }

    // wave-uniform output routing
    const int t    = nb / DIMV;          // 0=q 1=k 2=v
    const int head = (nb % DIMV) >> 6;
    const int b    = m0 / SEQ;
    const int q0   = m0 % SEQ;
    const int gh   = b * 12 + head;
    float bias[4];
#pragma unroll
    for (int ni = 0; ni < 4; ++ni) bias[ni] = qkv_b[nb + ni * 16 + n16];

    __bf16* dstQK = (t == 0) ? qbuf : kbuf;
#pragma unroll
    for (int mi = 0; mi < 2; ++mi)
#pragma unroll
        for (int ni = 0; ni < 4; ++ni)
#pragma unroll
            for (int r = 0; r < 8; ++r) {
                int qi = q0 + mi * 16 + r + 8 * hl;
                int c  = ni * 16 + n16;
                __bf16 bv = (__bf16)(acc[mi][ni][r] + bias[ni]);
                if (t < 2) dstQK[((size_t)(gh * SEQ + qi)) * HDV + c] = bv;
                else       vT[((size_t)(gh * HDV + c)) * SEQ + qi] = bv;
            }
}

// ---------------------------------------------------------------------------
// Kernel 2: decomposed relative-position bias GEMMs (one wave per 48x48 tile)
// ---------------------------------------------------------------------------
__global__ void __launch_bounds__(32) relbias_kernel(
    const __bf16* __restrict__ qbuf, const __bf16* __restrict__ relb_h,
    const __bf16* __restrict__ relb_w, float* __restrict__ relH,
    float* __restrict__ relW)
{
    const int lane = threadIdx.x & 31;
    const int hl = lane >> 4, n16 = lane & 15;
    const int gh = blockIdx.x;
    const bool isW = blockIdx.y >= 48;
    const int fixed = isW ? (blockIdx.y - 48) : blockIdx.y;
    const __bf16* rel = isW ? relb_w : relb_h;
    v8f acc[9] = {};

    for (int kc = 0; kc < HDV; kc += 32) {
        v16bf bfr[3], af[3];
#pragma unroll
        for (int ni = 0; ni < 3; ++ni) {
            int ridx = fixed - (16 * ni + n16) + 47;   // in [0, 94]
            const __bf16* rp = &rel[(size_t)ridx * HDV + kc + 16 * hl];
#pragma unroll
            for (int e = 0; e < 16; ++e) bfr[ni][e] = rp[e];
        }
#pragma unroll
        for (int mi = 0; mi < 3; ++mi) {
            int row  = 16 * mi + n16;
            int qidx = isW ? (row * 48 + fixed) : (fixed * 48 + row);
            const __bf16* qp = &qbuf[((size_t)(gh * SEQ + qidx)) * HDV + kc];
#pragma unroll
            for (int e = 0; e < 16; ++e) af[mi][e] = qp[a_k_idx(e, hl)];
        }
#pragma unroll
        for (int mi = 0; mi < 3; ++mi)
#pragma unroll
            for (int ni = 0; ni < 3; ++ni)
                acc[mi * 3 + ni] = wmma_bf16(af[mi], bfr[ni], acc[mi * 3 + ni]);
    }
    float* dst = isW ? relW : relH;
#pragma unroll
    for (int mi = 0; mi < 3; ++mi)
#pragma unroll
        for (int ni = 0; ni < 3; ++ni)
#pragma unroll
            for (int r = 0; r < 8; ++r) {
                int row  = 16 * mi + r + 8 * hl;
                int col  = 16 * ni + n16;
                int qidx = isW ? (row * 48 + fixed) : (fixed * 48 + row);
                dst[((size_t)(gh * SEQ + qidx)) * 48 + col] = acc[mi * 3 + ni][r];
            }
}

// ---------------------------------------------------------------------------
// Kernel 3: fused flash attention with decomposed rel-pos bias.
// K/V chunks (block-shared) staged via TDM into double-buffered LDS; DMA for
// chunk i+1 overlaps compute on chunk i. 8 waves x 16 query rows per block.
// ---------------------------------------------------------------------------
__global__ void __launch_bounds__(256) attn_flash_kernel(
    const __bf16* __restrict__ qbuf, const __bf16* __restrict__ kbuf,
    const __bf16* __restrict__ vT, const float* __restrict__ relH,
    const float* __restrict__ relW, __bf16* __restrict__ attn_out)
{
    __shared__ __bf16 ldsK[2][32][64];   // [buf][key][c]
    __shared__ __bf16 ldsV[2][64][32];   // [buf][c][key]
    __shared__ __bf16 ldsP[8][16][32];   // per-wave P staging (C->A relayout)
    const int lane = threadIdx.x & 31;
    const int wave = threadIdx.x >> 5;
    const int hl = lane >> 4, n16 = lane & 15;
    const int gh  = blockIdx.y;
    const int m_q = blockIdx.x * 128 + wave * 16;
    const float scale = 0.125f;          // hd^-0.5, hd=64

    auto stage = [&](int bb, int kbn) {
#if HAVE_TDM
        if (wave == 0) {
            tdm_load_2d_bf16(lds_addr_of(&ldsK[bb][0][0]),
                             &kbuf[((size_t)(gh * SEQ + kbn)) * HDV], HDV, 32, HDV);
            tdm_load_2d_bf16(lds_addr_of(&ldsV[bb][0][0]),
                             &vT[(size_t)gh * HDV * SEQ + kbn], 32, HDV, SEQ);
        }
#else
        {
            int t8 = threadIdx.x * 8;    // K: 2048 elems, 8 per thread
            const __bf16* ks = &kbuf[((size_t)(gh * SEQ + kbn)) * HDV];
#pragma unroll
            for (int j = 0; j < 8; ++j) (&ldsK[bb][0][0])[t8 + j] = ks[t8 + j];
            int vrow = threadIdx.x >> 2, vcol = (threadIdx.x & 3) * 8;
            const __bf16* vs = &vT[((size_t)(gh * HDV + vrow)) * SEQ + kbn + vcol];
#pragma unroll
            for (int j = 0; j < 8; ++j) ldsV[bb][vrow][vcol + j] = vs[j];
        }
#endif
    };
    auto commit = [&]() {
#if HAVE_TDM
        if (wave == 0) __builtin_amdgcn_s_wait_tensorcnt((short)0);
#endif
        __syncthreads();
    };

    // Q A-fragments (unscaled; scale applied to S post-WMMA)
    v16bf qa[2];
#pragma unroll
    for (int kc = 0; kc < 2; ++kc) {
        const __bf16* qp = &qbuf[((size_t)(gh * SEQ + m_q + n16)) * HDV + kc * 32];
#pragma unroll
        for (int e = 0; e < 16; ++e) qa[kc][e] = qp[a_k_idx(e, hl)];
    }

    v8f o[4] = {};
    float rm[8], rs[8];
#pragma unroll
    for (int r = 0; r < 8; ++r) { rm[r] = -1e30f; rs[r] = 0.f; }

    stage(0, 0);
    commit();
    for (int c = 0; c < SEQ / 32; ++c) {
        const int kb = c * 32, bb = c & 1;
        if (kb + 32 < SEQ) stage(bb ^ 1, kb + 32);

        // ---- fragments from LDS + bias from global (batched) ----
        v16bf kf[2][2];
#pragma unroll
        for (int t = 0; t < 2; ++t)
#pragma unroll
            for (int kc = 0; kc < 2; ++kc) {
                const __bf16* kp = &ldsK[bb][16 * t + n16][kc * 32 + 16 * hl];
#pragma unroll
                for (int e = 0; e < 16; ++e) kf[t][kc][e] = kp[e];
            }
        v16bf vf[4];
#pragma unroll
        for (int ni = 0; ni < 4; ++ni) {
            const __bf16* vp = &ldsV[bb][ni * 16 + n16][16 * hl];
#pragma unroll
            for (int e = 0; e < 16; ++e) vf[ni][e] = vp[e];
        }
        float biasv[2][8];
#pragma unroll
        for (int t = 0; t < 2; ++t) {
            int key = kb + 16 * t + n16;
            int kh = key / 48, kw = key % 48;
#pragma unroll
            for (int r = 0; r < 8; ++r) {
                size_t qoff = ((size_t)(gh * SEQ + m_q + r + 8 * hl)) * 48;
                biasv[t][r] = relH[qoff + kh] + relW[qoff + kw];
            }
        }
        // ---- S = scale * Q.K^T + bias ----
        float s[2][8];
#pragma unroll
        for (int t = 0; t < 2; ++t) {
            v8f sacc = {};
            sacc = wmma_bf16(qa[0], kf[t][0], sacc);
            sacc = wmma_bf16(qa[1], kf[t][1], sacc);
#pragma unroll
            for (int r = 0; r < 8; ++r) s[t][r] = sacc[r] * scale + biasv[t][r];
        }
        // ---- online softmax ----
        float corr[8];
#pragma unroll
        for (int r = 0; r < 8; ++r) {
            float lmax = fmaxf(s[0][r], s[1][r]);
#pragma unroll
            for (int m = 8; m >= 1; m >>= 1) lmax = fmaxf(lmax, __shfl_xor(lmax, m, 32));
            float mnew = fmaxf(rm[r], lmax);
            float p0 = __expf(s[0][r] - mnew);
            float p1 = __expf(s[1][r] - mnew);
            s[0][r] = p0; s[1][r] = p1;
            float lsum = p0 + p1;
#pragma unroll
            for (int m = 8; m >= 1; m >>= 1) lsum += __shfl_xor(lsum, m, 32);
            float cc = __expf(rm[r] - mnew);
            rs[r] = rs[r] * cc + lsum;
            rm[r] = mnew;
            corr[r] = cc;
        }
#pragma unroll
        for (int ni = 0; ni < 4; ++ni)
#pragma unroll
            for (int r = 0; r < 8; ++r) o[ni][r] *= corr[r];

        // ---- P: C-layout -> per-wave LDS tile -> A-fragment ----
        __builtin_amdgcn_wave_barrier();
#pragma unroll
        for (int t = 0; t < 2; ++t)
#pragma unroll
            for (int r = 0; r < 8; ++r)
                ldsP[wave][r + 8 * hl][16 * t + n16] = (__bf16)s[t][r];
        __builtin_amdgcn_wave_barrier();
        v16bf pa;
#pragma unroll
        for (int j = 0; j < 8; ++j) {
            int K0 = 2 * j + 16 * hl;
            pa[2 * j]     = ldsP[wave][n16][K0];
            pa[2 * j + 1] = ldsP[wave][n16][K0 + 1];
        }
        __builtin_amdgcn_wave_barrier();
#pragma unroll
        for (int ni = 0; ni < 4; ++ni) o[ni] = wmma_bf16(pa, vf[ni], o[ni]);

        commit();
    }
    // normalize + scatter to (b, n, nh*hd) layout
    const int b = gh / 12, h = gh % 12;
#pragma unroll
    for (int ni = 0; ni < 4; ++ni)
#pragma unroll
        for (int r = 0; r < 8; ++r) {
            int qrow = m_q + r + 8 * hl;
            int c = ni * 16 + n16;
            float val = o[ni][r] / rs[r];
            attn_out[((size_t)(b * SEQ + qrow)) * DIMV + h * HDV + c] = (__bf16)val;
        }
}

// ---------------------------------------------------------------------------
// Kernel 4: output projection GEMM (4608x768 bf16)@proj_w^T+proj_b -> f32.
// Same TDM-staged-B structure as kernel 1.
// ---------------------------------------------------------------------------
__global__ void __launch_bounds__(256) proj_gemm_kernel(
    const __bf16* __restrict__ a_in, const __bf16* __restrict__ pwb,
    const float* __restrict__ proj_b, float* __restrict__ out)
{
    __shared__ __bf16 ldsB[2][64][32];
    const int lane = threadIdx.x & 31, wave = threadIdx.x >> 5;
    const int hl = lane >> 4, n16 = lane & 15;
    const int m0 = blockIdx.x * 256 + wave * 32;
    const int nb = blockIdx.y * 64;
    v8f acc[2][4] = {};

    auto stageB = [&](int bb, int kc) {
#if HAVE_TDM
        if (wave == 0)
            tdm_load_2d_bf16(lds_addr_of(&ldsB[bb][0][0]),
                             &pwb[(size_t)nb * DIMV + kc], 32, 64, DIMV);
#else
        int brow = threadIdx.x >> 2, bcol = (threadIdx.x & 3) * 8;
        const __bf16* bs = &pwb[(size_t)(nb + brow) * DIMV + kc + bcol];
#pragma unroll
        for (int j = 0; j < 8; ++j) ldsB[bb][brow][bcol + j] = bs[j];
#endif
    };
    auto commit = [&]() {
#if HAVE_TDM
        if (wave == 0) __builtin_amdgcn_s_wait_tensorcnt((short)0);
#endif
        __syncthreads();
    };

    stageB(0, 0);
    commit();
    for (int c = 0; c < DIMV / 32; ++c) {
        const int kc = c * 32, bb = c & 1;
        if (kc + 32 < DIMV) stageB(bb ^ 1, kc + 32);
        v16bf af[2], bfr[4];
#pragma unroll
        for (int mi = 0; mi < 2; ++mi) {
            const __bf16* ap = &a_in[(size_t)(m0 + mi * 16 + n16) * DIMV + kc];
#pragma unroll
            for (int e = 0; e < 16; ++e) af[mi][e] = ap[a_k_idx(e, hl)];
        }
#pragma unroll
        for (int ni = 0; ni < 4; ++ni) {
            const __bf16* bp = &ldsB[bb][ni * 16 + n16][16 * hl];
#pragma unroll
            for (int e = 0; e < 16; ++e) bfr[ni][e] = bp[e];
        }
#pragma unroll
        for (int mi = 0; mi < 2; ++mi)
#pragma unroll
            for (int ni = 0; ni < 4; ++ni)
                acc[mi][ni] = wmma_bf16(af[mi], bfr[ni], acc[mi][ni]);
        commit();
    }
    float bias[4];
#pragma unroll
    for (int ni = 0; ni < 4; ++ni) bias[ni] = proj_b[nb + ni * 16 + n16];
#pragma unroll
    for (int mi = 0; mi < 2; ++mi)
#pragma unroll
        for (int ni = 0; ni < 4; ++ni)
#pragma unroll
            for (int r = 0; r < 8; ++r) {
                int m = m0 + mi * 16 + r + 8 * hl;
                out[(size_t)m * DIMV + nb + ni * 16 + n16] = acc[mi][ni][r] + bias[ni];
            }
}

// ---------------------------------------------------------------------------
extern "C" void kernel_launch(void* const* d_in, const int* in_sizes, int n_in,
                              void* d_out, int out_size, void* d_ws, size_t ws_size,
                              hipStream_t stream) {
    const float* x         = (const float*)d_in[0];
    const float* rel_pos_h = (const float*)d_in[1];
    const float* rel_pos_w = (const float*)d_in[2];
    const float* qkv_w     = (const float*)d_in[3];
    const float* qkv_b     = (const float*)d_in[4];
    const float* proj_w    = (const float*)d_in[5];
    const float* proj_b    = (const float*)d_in[6];
    float* out = (float*)d_out;

    char* ws = (char*)d_ws;
    size_t off = 0;
    auto alloc = [&](size_t bytes) {
        char* p = ws + off;
        off += (bytes + 255) & ~(size_t)255;
        return p;
    };
    const size_t qkv_elems = (size_t)NHEADS * SEQ * HDV;
    __bf16* qbuf  = (__bf16*)alloc(qkv_elems * sizeof(__bf16));
    __bf16* kbuf  = (__bf16*)alloc(qkv_elems * sizeof(__bf16));
    __bf16* vT    = (__bf16*)alloc(qkv_elems * sizeof(__bf16));
    float*  relH  = (float*)alloc((size_t)NHEADS * SEQ * 48 * sizeof(float));
    float*  relW  = (float*)alloc((size_t)NHEADS * SEQ * 48 * sizeof(float));
    __bf16* attnb = (__bf16*)alloc((size_t)MROWS * DIMV * sizeof(__bf16));
    __bf16* xb    = (__bf16*)alloc((size_t)MROWS * DIMV * sizeof(__bf16));
    __bf16* wb    = (__bf16*)alloc((size_t)3 * DIMV * DIMV * sizeof(__bf16));
    __bf16* pwb   = (__bf16*)alloc((size_t)DIMV * DIMV * sizeof(__bf16));
    __bf16* rbh   = (__bf16*)alloc((size_t)95 * HDV * sizeof(__bf16));
    __bf16* rbw   = (__bf16*)alloc((size_t)95 * HDV * sizeof(__bf16));

    auto pack = [&](const float* s, __bf16* d, int n) {
        int blocks = (n / 8 + 255) / 256;
        pack_bf16_kernel<<<blocks, 256, 0, stream>>>(s, d, n);
    };
    pack(x, xb, MROWS * DIMV);
    pack(qkv_w, wb, 3 * DIMV * DIMV);
    pack(proj_w, pwb, DIMV * DIMV);
    pack(rel_pos_h, rbh, 95 * HDV);
    pack(rel_pos_w, rbw, 95 * HDV);

    qkv_gemm_kernel<<<dim3(MROWS / 256, (3 * DIMV) / 64), 256, 0, stream>>>(
        xb, wb, qkv_b, qbuf, kbuf, vT);
    relbias_kernel<<<dim3(NHEADS, 96), 32, 0, stream>>>(
        qbuf, rbh, rbw, relH, relW);
    attn_flash_kernel<<<dim3(SEQ / 128, NHEADS), 256, 0, stream>>>(
        qbuf, kbuf, vT, relH, relW, attnb);
    proj_gemm_kernel<<<dim3(MROWS / 256, DIMV / 64), 256, 0, stream>>>(
        attnb, pwb, proj_b, out);
}
